// MixtralMLP_23072564314325
// MI455X (gfx1250) — matrix operand order repaired and checked by reading, not compiled, not run on hardware
//
#include <hip/hip_runtime.h>

#define D_MODEL 4096
#define D_FF    14336
#define T_TOK   8192

#define BK 32        // k chunk (WMMA K)
#define GU_BM 256    // gate/up token tile
#define GU_BN 64     // gate/up ff tile
#define DN_BM 256    // down token tile
#define DN_BN 128    // down d_model tile

typedef __attribute__((ext_vector_type(16))) __bf16 bf16x16;
typedef __attribute__((ext_vector_type(8)))  __bf16 bf16x8;
typedef __attribute__((ext_vector_type(4)))  __bf16 bf16x4;
typedef __attribute__((ext_vector_type(8)))  float  f32x8;
typedef int v4i __attribute__((vector_size(16)));

#if __has_builtin(__builtin_amdgcn_global_load_async_to_lds_b128)
#define HAVE_ASYNC 1
#else
#define HAVE_ASYNC 0
#endif

// ---------------------------------------------------------------------------
// Small helpers
// ---------------------------------------------------------------------------
__device__ __forceinline__ void wait_async0() {
#if __has_builtin(__builtin_amdgcn_s_wait_asynccnt)
  __builtin_amdgcn_s_wait_asynccnt(0);
#elif HAVE_ASYNC
  asm volatile("s_wait_asynccnt 0x0" ::: "memory");
#endif
}

#if HAVE_ASYNC
typedef __attribute__((address_space(1))) v4i* gv4i_p;
typedef __attribute__((address_space(3))) v4i* lv4i_p;
__device__ __forceinline__ void async_cp16(const __bf16* gsrc, __bf16* ldst) {
  __builtin_amdgcn_global_load_async_to_lds_b128(
      (gv4i_p)(__bf16*)gsrc, (lv4i_p)ldst, 0, 0);
}
#endif

__device__ __forceinline__ f32x8 wmma_bf16(bf16x16 a, bf16x16 b, f32x8 c) {
  return __builtin_amdgcn_wmma_f32_16x16x32_bf16(
      false, a, false, b, (short)0, c, false, false);
}

// A fragment (16x32 bf16) from LDS tile stored [m][k], k contiguous (stride BK).
// lanes 0-15 -> M=lane, K=0..15; lanes 16-31 -> M=lane-16, K=16..31.
__device__ __forceinline__ bf16x16 load_frag_A(const __bf16* base, int mrow, int lane) {
  const int half = lane >> 4;
  const __bf16* p = base + (mrow + (lane & 15)) * BK + half * 16;
  bf16x16 a;
#pragma unroll
  for (int j = 0; j < 8; ++j) a[j] = p[j];
#pragma unroll
  for (int j = 0; j < 8; ++j) a[8 + j] = p[8 + j];
  return a;
}

// B fragment (32x16 bf16) from LDS tile stored [n][k], k contiguous.
// lanes 0-15 -> N=lane, K=0..15; lanes 16-31 -> N=lane-16, K=16..31.
__device__ __forceinline__ bf16x16 load_frag_B(const __bf16* base, int ncol, int lane) {
  const int half = lane >> 4;
  const __bf16* p = base + (ncol + (lane & 15)) * BK + half * 16;
  bf16x16 b;
#pragma unroll
  for (int j = 0; j < 16; ++j) b[j] = p[j];
  return b;
}

// ---------------------------------------------------------------------------
// fp32 -> bf16 bulk convert
// ---------------------------------------------------------------------------
__global__ void cvt_f32_bf16_kernel(const float* __restrict__ src,
                                    __bf16* __restrict__ dst, int n4) {
  int i = blockIdx.x * blockDim.x + threadIdx.x;
  int stride = gridDim.x * blockDim.x;
  for (int q = i; q < n4; q += stride) {
    float4 v = reinterpret_cast<const float4*>(src)[q];
    bf16x4 o;
    o[0] = (__bf16)v.x; o[1] = (__bf16)v.y;
    o[2] = (__bf16)v.z; o[3] = (__bf16)v.w;
    reinterpret_cast<bf16x4*>(dst)[q] = o;
  }
}

// ---------------------------------------------------------------------------
// fp32 [R][C] -> bf16 transposed [C][R]  (64x64 LDS tile, coalesced both ways)
// ---------------------------------------------------------------------------
#define TT 64
__global__ __launch_bounds__(256) void tcvt_kernel(const float* __restrict__ src,
                                                   __bf16* __restrict__ dst,
                                                   int R, int C) {
  __shared__ __bf16 tile[TT][TT + 8];
  const int c0 = blockIdx.x * TT, r0 = blockIdx.y * TT;
  const int tid = threadIdx.x;
#pragma unroll
  for (int s = 0; s < 4; ++s) {
    int q = tid + s * 256;
    int r = q >> 4, cc = (q & 15) * 4;
    float4 v = *reinterpret_cast<const float4*>(src + (long)(r0 + r) * C + c0 + cc);
    tile[cc + 0][r] = (__bf16)v.x;
    tile[cc + 1][r] = (__bf16)v.y;
    tile[cc + 2][r] = (__bf16)v.z;
    tile[cc + 3][r] = (__bf16)v.w;
  }
  __syncthreads();
#pragma unroll
  for (int s = 0; s < 4; ++s) {
    int q = tid + s * 256;
    int c = q >> 2, rr = (q & 3) * 16;
    __bf16* dp = dst + (long)(c0 + c) * R + r0 + rr;
#pragma unroll
    for (int e = 0; e < 16; ++e) dp[e] = tile[c][rr + e];
  }
}

// ---------------------------------------------------------------------------
// Kernel 1: fused gate/up GEMM + SwiGLU  ->  H (bf16) [T, D_FF]
//   A = Xb [T][D_MODEL] bf16, B = Wgt/Wut [D_FF][D_MODEL] bf16 (pre-transposed)
//   grid.x = T/GU_BM (fast), grid.y = D_FF/GU_BN
//   8 waves as 4(M) x 2(N); wave tile 64x32 per matrix -> 16 WMMA / k-step
// ---------------------------------------------------------------------------
__global__ __launch_bounds__(256) void gateup_kernel(
    const __bf16* __restrict__ Xb, const __bf16* __restrict__ Wgt,
    const __bf16* __restrict__ Wut, __bf16* __restrict__ Hb)
{
  __shared__ alignas(16) __bf16 Xs[2][GU_BM * BK];   // 2 x 16 KiB
  __shared__ alignas(16) __bf16 Gs[2][GU_BN * BK];   // 2 x  4 KiB
  __shared__ alignas(16) __bf16 Us[2][GU_BN * BK];   // 2 x  4 KiB

  const int tid  = threadIdx.x;
  const int lane = tid & 31;
  const int wave = tid >> 5;
  const int wm   = wave >> 1;               // 0..3 : 64-row band
  const int wn   = wave & 1;                // 0..1 : 32-col band
  const long m0  = (long)blockIdx.x * GU_BM;
  const long n0  = (long)blockIdx.y * GU_BN;

  f32x8 accG[4][2], accU[4][2];
  const f32x8 z = {0.f,0.f,0.f,0.f,0.f,0.f,0.f,0.f};
#pragma unroll
  for (int i = 0; i < 4; ++i)
#pragma unroll
    for (int j = 0; j < 2; ++j) { accG[i][j] = z; accU[i][j] = z; }

#if !HAVE_ASYNC
  uint4 xr[4], gr, ur;
#endif

  auto load_tiles = [&](int sel, int kt) {
    const long k0 = (long)kt * BK;
#if HAVE_ASYNC
#pragma unroll
    for (int s = 0; s < 4; ++s) {             // A: 1024 x 16B chunks
      int c = tid + s * 256;
      int row = c >> 2, pc = (c & 3) * 8;
      async_cp16(Xb + (m0 + row) * (long)D_MODEL + k0 + pc, &Xs[sel][row * BK + pc]);
    }
    {                                          // G/U: 256 x 16B chunks each
      int row = tid >> 2, pc = (tid & 3) * 8;
      async_cp16(Wgt + (n0 + row) * (long)D_MODEL + k0 + pc, &Gs[sel][row * BK + pc]);
      async_cp16(Wut + (n0 + row) * (long)D_MODEL + k0 + pc, &Us[sel][row * BK + pc]);
    }
#else
#pragma unroll
    for (int s = 0; s < 4; ++s) {
      int c = tid + s * 256;
      int row = c >> 2, pc = (c & 3) * 8;
      xr[s] = *reinterpret_cast<const uint4*>(Xb + (m0 + row) * (long)D_MODEL + k0 + pc);
    }
    {
      int row = tid >> 2, pc = (tid & 3) * 8;
      gr = *reinterpret_cast<const uint4*>(Wgt + (n0 + row) * (long)D_MODEL + k0 + pc);
      ur = *reinterpret_cast<const uint4*>(Wut + (n0 + row) * (long)D_MODEL + k0 + pc);
    }
#endif
  };

#if !HAVE_ASYNC
  auto commit_tiles = [&](int sel) {
#pragma unroll
    for (int s = 0; s < 4; ++s) {
      int c = tid + s * 256;
      int row = c >> 2, pc = (c & 3) * 8;
      *reinterpret_cast<uint4*>(&Xs[sel][row * BK + pc]) = xr[s];
    }
    {
      int row = tid >> 2, pc = (tid & 3) * 8;
      *reinterpret_cast<uint4*>(&Gs[sel][row * BK + pc]) = gr;
      *reinterpret_cast<uint4*>(&Us[sel][row * BK + pc]) = ur;
    }
  };
#endif

  // prologue: fill buffer 0
  load_tiles(0, 0);
#if HAVE_ASYNC
  wait_async0();
#else
  commit_tiles(0);
#endif
  __syncthreads();

  const int KT = D_MODEL / BK;   // 128
  for (int kt = 0; kt < KT; ++kt) {
    const int cur = kt & 1;
    const bool more = (kt + 1 < KT);
    if (more) load_tiles(1 - cur, kt + 1);   // fills other buffer, overlaps WMMA

    const __bf16* Xp = Xs[cur];
    const __bf16* Gp = Gs[cur];
    const __bf16* Up = Us[cur];

    bf16x16 a[4];
#pragma unroll
    for (int i = 0; i < 4; ++i) a[i] = load_frag_A(Xp, wm * 64 + i * 16, lane);
    bf16x16 bg[2], bu[2];
#pragma unroll
    for (int j = 0; j < 2; ++j) {
      bg[j] = load_frag_B(Gp, wn * 32 + j * 16, lane);
      bu[j] = load_frag_B(Up, wn * 32 + j * 16, lane);
    }
#pragma unroll
    for (int i = 0; i < 4; ++i)
#pragma unroll
      for (int j = 0; j < 2; ++j) {
        accG[i][j] = wmma_bf16(a[i], bg[j], accG[i][j]);
        accU[i][j] = wmma_bf16(a[i], bu[j], accU[i][j]);
      }

    if (more) {
#if HAVE_ASYNC
      wait_async0();
#else
      commit_tiles(1 - cur);
#endif
      __syncthreads();
    }
  }

  // Epilogue: h = silu(gate) * up, store bf16
  const int half = lane >> 4, nlo = lane & 15;
#pragma unroll
  for (int i = 0; i < 4; ++i)
#pragma unroll
    for (int j = 0; j < 2; ++j) {
      f32x8 g = accG[i][j], u = accU[i][j];
#pragma unroll
      for (int e = 0; e < 8; ++e) {
        float gv = g[e];
        float hv = (gv / (1.f + __expf(-gv))) * u[e];
        long row = m0 + wm * 64 + i * 16 + half * 8 + e;
        long col = n0 + wn * 32 + j * 16 + nlo;
        Hb[row * D_FF + col] = (__bf16)hv;
      }
    }
}

// ---------------------------------------------------------------------------
// Kernel 2: down projection  out = routing * (H · Wd)
//   A = Hb [T][D_FF] bf16, B = Wdt [D_MODEL][D_FF] bf16 (pre-transposed)
//   grid.x = D_MODEL/DN_BN (fast: Wdt 112 MiB stays L2-resident), grid.y = T/DN_BM
//   8 waves as 4(M) x 2(N); wave tile 64x64 -> 16 WMMA / k-step
// ---------------------------------------------------------------------------
__global__ __launch_bounds__(256) void down_kernel(
    const __bf16* __restrict__ Hb, const __bf16* __restrict__ Wdt,
    const float* __restrict__ Rw, float* __restrict__ Out)
{
  __shared__ alignas(16) __bf16 Hs[2][DN_BM * BK];   // 2 x 16 KiB
  __shared__ alignas(16) __bf16 Ds[2][DN_BN * BK];   // 2 x  8 KiB
  __shared__ float Rs[DN_BM];

  const int tid  = threadIdx.x;
  const int lane = tid & 31;
  const int wave = tid >> 5;
  const int wm   = wave >> 1;               // 0..3 : 64-row band
  const int wn   = wave & 1;                // 0..1 : 64-col band
  const long n0  = (long)blockIdx.x * DN_BN;
  const long m0  = (long)blockIdx.y * DN_BM;

  Rs[tid] = Rw[m0 + tid];   // DN_BM == 256 == blockDim.x

  f32x8 acc[4][4];
  const f32x8 z = {0.f,0.f,0.f,0.f,0.f,0.f,0.f,0.f};
#pragma unroll
  for (int i = 0; i < 4; ++i)
#pragma unroll
    for (int j = 0; j < 4; ++j) acc[i][j] = z;

#if !HAVE_ASYNC
  uint4 hr[4], dr[2];
#endif

  auto load_tiles = [&](int sel, int kt) {
    const long k0 = (long)kt * BK;
#if HAVE_ASYNC
#pragma unroll
    for (int s = 0; s < 4; ++s) {             // A: 1024 chunks
      int c = tid + s * 256;
      int row = c >> 2, pc = (c & 3) * 8;
      async_cp16(Hb + (m0 + row) * (long)D_FF + k0 + pc, &Hs[sel][row * BK + pc]);
    }
#pragma unroll
    for (int s = 0; s < 2; ++s) {             // B: 512 chunks
      int c = tid + s * 256;
      int row = c >> 2, pc = (c & 3) * 8;
      async_cp16(Wdt + (n0 + row) * (long)D_FF + k0 + pc, &Ds[sel][row * BK + pc]);
    }
#else
#pragma unroll
    for (int s = 0; s < 4; ++s) {
      int c = tid + s * 256;
      int row = c >> 2, pc = (c & 3) * 8;
      hr[s] = *reinterpret_cast<const uint4*>(Hb + (m0 + row) * (long)D_FF + k0 + pc);
    }
#pragma unroll
    for (int s = 0; s < 2; ++s) {
      int c = tid + s * 256;
      int row = c >> 2, pc = (c & 3) * 8;
      dr[s] = *reinterpret_cast<const uint4*>(Wdt + (n0 + row) * (long)D_FF + k0 + pc);
    }
#endif
  };

#if !HAVE_ASYNC
  auto commit_tiles = [&](int sel) {
#pragma unroll
    for (int s = 0; s < 4; ++s) {
      int c = tid + s * 256;
      int row = c >> 2, pc = (c & 3) * 8;
      *reinterpret_cast<uint4*>(&Hs[sel][row * BK + pc]) = hr[s];
    }
#pragma unroll
    for (int s = 0; s < 2; ++s) {
      int c = tid + s * 256;
      int row = c >> 2, pc = (c & 3) * 8;
      *reinterpret_cast<uint4*>(&Ds[sel][row * BK + pc]) = dr[s];
    }
  };
#endif

  load_tiles(0, 0);
#if HAVE_ASYNC
  wait_async0();
#else
  commit_tiles(0);
#endif
  __syncthreads();

  const int KT = D_FF / BK;   // 448
  for (int kt = 0; kt < KT; ++kt) {
    const int cur = kt & 1;
    const bool more = (kt + 1 < KT);
    if (more) load_tiles(1 - cur, kt + 1);

    const __bf16* Hp = Hs[cur];
    const __bf16* Dp = Ds[cur];

    bf16x16 a[4];
#pragma unroll
    for (int i = 0; i < 4; ++i) a[i] = load_frag_A(Hp, wm * 64 + i * 16, lane);
    bf16x16 b[4];
#pragma unroll
    for (int j = 0; j < 4; ++j) b[j] = load_frag_B(Dp, wn * 64 + j * 16, lane);

#pragma unroll
    for (int i = 0; i < 4; ++i)
#pragma unroll
      for (int j = 0; j < 4; ++j)
        acc[i][j] = wmma_bf16(a[i], b[j], acc[i][j]);

    if (more) {
#if HAVE_ASYNC
      wait_async0();
#else
      commit_tiles(1 - cur);
#endif
      __syncthreads();
    }
  }

  const int half = lane >> 4, nlo = lane & 15;
#pragma unroll
  for (int i = 0; i < 4; ++i)
#pragma unroll
    for (int j = 0; j < 4; ++j) {
      f32x8 a = acc[i][j];
#pragma unroll
      for (int e = 0; e < 8; ++e) {
        int lrow = wm * 64 + i * 16 + half * 8 + e;
        long row = m0 + lrow;
        long col = n0 + wn * 64 + j * 16 + nlo;
        Out[row * D_MODEL + col] = Rs[lrow] * a[e];
      }
    }
}

// ---------------------------------------------------------------------------
// Launcher
// ---------------------------------------------------------------------------
extern "C" void kernel_launch(void* const* d_in, const int* in_sizes, int n_in,
                              void* d_out, int out_size, void* d_ws, size_t ws_size,
                              hipStream_t stream) {
  const float* X  = (const float*)d_in[0];   // [T, D_MODEL]
  const float* Rw = (const float*)d_in[1];   // [T, 1]
  const float* Wg = (const float*)d_in[2];   // [D_MODEL, D_FF]
  const float* Wu = (const float*)d_in[3];   // [D_MODEL, D_FF]
  const float* Wd = (const float*)d_in[4];   // [D_FF, D_MODEL]
  float* Out = (float*)d_out;

  char* ws = (char*)d_ws;
  const size_t xb_bytes = (size_t)T_TOK * D_MODEL * sizeof(__bf16);  //  64 MiB
  const size_t w_bytes  = (size_t)D_FF * D_MODEL * sizeof(__bf16);   // 112 MiB
  __bf16* Xb  = (__bf16*)(ws);
  __bf16* Wgt = (__bf16*)(ws + xb_bytes);                  // [D_FF][D_MODEL]
  __bf16* Wut = (__bf16*)(ws + xb_bytes + w_bytes);        // [D_FF][D_MODEL]
  __bf16* Wdt = (__bf16*)(ws + xb_bytes + 2 * w_bytes);    // [D_MODEL][D_FF]
  __bf16* Hb  = (__bf16*)(ws + xb_bytes + 3 * w_bytes);    // [T][D_FF]

  // 1) one-time bf16 conversions (weights also transposed to [n][k])
  cvt_f32_bf16_kernel<<<4096, 256, 0, stream>>>(X, Xb, T_TOK * D_MODEL / 4);
  {
    dim3 g(D_FF / TT, D_MODEL / TT);   // Wg,Wu: [4096][14336] -> [14336][4096]
    tcvt_kernel<<<g, 256, 0, stream>>>(Wg, Wgt, D_MODEL, D_FF);
    tcvt_kernel<<<g, 256, 0, stream>>>(Wu, Wut, D_MODEL, D_FF);
  }
  {
    dim3 g(D_MODEL / TT, D_FF / TT);   // Wd: [14336][4096] -> [4096][14336]
    tcvt_kernel<<<g, 256, 0, stream>>>(Wd, Wdt, D_FF, D_MODEL);
  }

  // 2) fused gate/up + SwiGLU -> H
  dim3 gA(T_TOK / GU_BM, D_FF / GU_BN);   // x fast: weight stripe shared via L2
  gateup_kernel<<<gA, 256, 0, stream>>>(Xb, Wgt, Wut, Hb);

  // 3) down projection + routing scale
  dim3 gB(D_MODEL / DN_BN, T_TOK / DN_BM); // x fast: Wdt stays L2-resident
  down_kernel<<<gB, 256, 0, stream>>>(Hb, Wdt, Rw, Out);
}
